// ContextualBlock_37812892074157
// MI455X (gfx1250) — matrix operand order repaired and checked by reading, not compiled, not run on hardware
//
#include <hip/hip_runtime.h>

// ---------------- problem constants ----------------
#define BB     2
#define HH     64
#define WW     64
#define DD     64
#define KK     3
#define PDIM   576              // K*K*D
#define PH     62
#define CREAL  3844             // 62*62
#define CPAD   3872             // 121 * 32  (k-padding for bf16 WMMA, also 242*16)
#define MTOT   4096             // H*W
#define LAMDA  10.0f

typedef __attribute__((ext_vector_type(16))) __bf16 v16bf;
typedef __attribute__((ext_vector_type(8)))  __bf16 v8bf;
typedef __attribute__((ext_vector_type(8)))  float  v8f;

__device__ __forceinline__ __bf16 f2bf(float f) {
    unsigned u = __builtin_bit_cast(unsigned, f);
    unsigned r = (u + 0x7FFFu + ((u >> 16) & 1u)) >> 16;   // round-to-nearest-even
    unsigned short s = (unsigned short)r;
    return __builtin_bit_cast(__bf16, s);
}
__device__ __forceinline__ float bf2f(__bf16 b) {
    unsigned short s = __builtin_bit_cast(unsigned short, b);
    unsigned u = ((unsigned)s) << 16;
    return __builtin_bit_cast(float, u);
}

__device__ __forceinline__ v16bf joinbf(v8bf lo, v8bf hi) {
    v16bf r;
#pragma unroll
    for (int i = 0; i < 8; ++i) { r[i] = lo[i]; r[i + 8] = hi[i]; }
    return r;
}

// ---------------- prep: fg patches (A of GEMM1) + wwd ----------------
__global__ __launch_bounds__(64) void prep_fg_kernel(const float* __restrict__ fg,
                                                     __bf16* __restrict__ patch2,
                                                     float* __restrict__ wwd) {
    int bm = blockIdx.x;                 // b*4096 + m
    int ch = threadIdx.x;                // 0..63
    int b  = bm >> 12, m = bm & 4095;
    int y  = m >> 6,  x = m & 63;
    __shared__ float red[64];
    float ls = 0.f;
    __bf16* dst = patch2 + (size_t)bm * PDIM;
#pragma unroll
    for (int i = 0; i < 3; ++i)
#pragma unroll
        for (int j = 0; j < 3; ++j) {
            int yy = y + i - 1, xx = x + j - 1;
            float v = 0.f;
            if (yy >= 0 && yy < HH && xx >= 0 && xx < WW)
                v = fg[(((size_t)(b * HH + yy) * WW + xx) * DD) + ch];
            dst[(i * 3 + j) * DD + ch] = f2bf(v);
            ls += v * v;
        }
    red[ch] = ls;
    __syncthreads();
    for (int st = 32; st > 0; st >>= 1) {
        if (ch < st) red[ch] += red[ch + st];
        __syncthreads();
    }
    if (ch == 0) wwd[bm] = red[0];
}

// ---------------- prep: bg patches (B of GEMM1 / B^T of GEMM2) + k1d ----------------
__global__ __launch_bounds__(64) void prep_bg_kernel(const float* __restrict__ bg_in,
                                                     const float* __restrict__ mask,
                                                     __bf16* __restrict__ patch1,
                                                     __bf16* __restrict__ patch1T,
                                                     float* __restrict__ k1d) {
    int bc = blockIdx.x;                 // b*CPAD + c
    int ch = threadIdx.x;
    int b  = bc / CPAD, c = bc % CPAD;
    __shared__ float red[64];
    float ls = 0.f;
    __bf16* dst = patch1 + (size_t)bc * PDIM;
    __bf16* dT  = patch1T + (size_t)b * PDIM * CPAD + c;
    if (c < CREAL) {
        int py = c / PH, px = c % PH;     // VALID patch origin
#pragma unroll
        for (int i = 0; i < 3; ++i)
#pragma unroll
            for (int j = 0; j < 3; ++j) {
                int yy = py + i, xx = px + j;
                size_t base = ((size_t)(b * HH + yy) * WW + xx) * DD;
                float v = bg_in[base + ch] * mask[base];   // channel-0 mask broadcast
                int p = (i * 3 + j) * DD + ch;
                __bf16 bv = f2bf(v);
                dst[p] = bv;
                dT[(size_t)p * CPAD] = bv;
                ls += v * v;
            }
    } else {
        __bf16 z = f2bf(0.f);
#pragma unroll
        for (int i = 0; i < 9; ++i) {
            int p = i * DD + ch;
            dst[p] = z;
            dT[(size_t)p * CPAD] = z;
        }
    }
    red[ch] = ls;
    __syncthreads();
    for (int st = 32; st > 0; st >>= 1) {
        if (ch < st) red[ch] += red[ch + st];
        __syncthreads();
    }
    if (ch == 0) k1d[bc] = red[0];
}

// ---------------- prep: W_ml transpose to bf16 [o][128] ----------------
__global__ void prep_w_kernel(const float* __restrict__ W, __bf16* __restrict__ wt) {
    int i = blockIdx.x * 256 + threadIdx.x;
    if (i >= 64 * 128) return;
    int o = i >> 7, ch = i & 127;
    wt[o * 128 + ch] = f2bf(W[(size_t)ch * 64 + o]);
}

// ---------------- GEMM1: DS1 = k1d + wwd - 2 * patch2 @ patch1^T (bf16 WMMA) ----------------
__global__ __launch_bounds__(128) void gemm1_kernel(const __bf16* __restrict__ patch2,
                                                    const __bf16* __restrict__ patch1,
                                                    const float* __restrict__ k1d,
                                                    const float* __restrict__ wwd,
                                                    __bf16* __restrict__ DS) {
    int mtg = blockIdx.x;                // b*256 + mtile
    int b = mtg >> 8, mt = mtg & 255;
    int tid = threadIdx.x, lane = tid & 31, wv = tid >> 5;
    int r = lane & 15, kh = lane >> 4;

    __shared__ __bf16 a1[16 * PDIM];
    __shared__ float  wloc[16];

    // Stage the 16x576 bf16 A-tile via the CDNA5 async global->LDS pipe
    // (GLOBAL_LOAD_ASYNC_TO_LDS_B128, tracked by ASYNCcnt): no VGPR round-trip.
    {
        const __bf16* p2b = patch2 + (size_t)(b * MTOT + mt * 16) * PDIM;
        for (int i = tid; i < 16 * PDIM / 8; i += 128) {
            unsigned lds_off = (unsigned)(size_t)(&a1[i * 8]);           // allocation-relative LDS addr
            unsigned long long ga = (unsigned long long)(size_t)(p2b + i * 8);
            asm volatile("global_load_async_to_lds_b128 %0, %1, off"
                         :: "v"(lds_off), "v"(ga) : "memory");
        }
        asm volatile("s_wait_asynccnt 0x0" ::: "memory");
    }
    if (tid < 16) wloc[tid] = wwd[b * MTOT + mt * 16 + tid];
    __syncthreads();

    int t = blockIdx.y * 4 + wv;         // n-tile (16 bg patches)
    if (t >= CPAD / 16) return;

    v8f acc = {0.f, 0.f, 0.f, 0.f, 0.f, 0.f, 0.f, 0.f};
    const __bf16* brow = patch1 + ((size_t)b * CPAD + t * 16 + r) * PDIM + kh * 8;
    const __bf16* arow = a1 + r * PDIM + kh * 8;
#pragma unroll
    for (int kt = 0; kt < PDIM / 32; ++kt) {   // 18 WMMAs
        v8bf alo = *(const v8bf*)(arow + kt * 32);
        v8bf ahi = *(const v8bf*)(arow + kt * 32 + 16);
        v8bf blo = *(const v8bf*)(brow + kt * 32);
        v8bf bhi = *(const v8bf*)(brow + kt * 32 + 16);
        acc = __builtin_amdgcn_wmma_f32_16x16x32_bf16(false, joinbf(alo, ahi),
                                                      false, joinbf(blo, bhi),
                                                      (short)0, acc, false, false);
    }
    int c = t * 16 + r;
    float kv = k1d[b * CPAD + c];
    __bf16* drow = DS + (size_t)(b * MTOT + mt * 16) * CPAD + c;
#pragma unroll
    for (int v = 0; v < 8; ++v) {
        int ml = v + 8 * kh;
        float ds = kv + wloc[ml] - 2.0f * acc[v];
        drow[(size_t)ml * CPAD] = f2bf(ds);
    }
}

// ---------------- row stats: mu, 1/sd (population std) ----------------
__global__ __launch_bounds__(256) void stats_kernel(const __bf16* __restrict__ DS,
                                                    float* __restrict__ mu,
                                                    float* __restrict__ rsd) {
    int row = blockIdx.x;
    const __bf16* d = DS + (size_t)row * CPAD;
    float s = 0.f, q = 0.f;
    for (int c = threadIdx.x; c < CREAL; c += 256) {
        float v = bf2f(d[c]);
        s += v; q += v * v;
    }
    __shared__ float ss[256], sq[256];
    ss[threadIdx.x] = s; sq[threadIdx.x] = q;
    __syncthreads();
    for (int st = 128; st > 0; st >>= 1) {
        if (threadIdx.x < st) { ss[threadIdx.x] += ss[threadIdx.x + st]; sq[threadIdx.x] += sq[threadIdx.x + st]; }
        __syncthreads();
    }
    if (threadIdx.x == 0) {
        float m = ss[0] / (float)CREAL;
        float var = sq[0] / (float)CREAL - m * m;
        mu[row] = m;
        rsd[row] = 1.0f / sqrtf(fmaxf(var, 1e-12f));
    }
}

// ---------------- CA = exp(-LAMDA*tanh((DS-mu)/sd)) in place; inv9 = 1/(9*sum) ----------------
__global__ __launch_bounds__(256) void expca_kernel(__bf16* __restrict__ DS,
                                                    const float* __restrict__ mu,
                                                    const float* __restrict__ rsd,
                                                    float* __restrict__ inv9) {
    int row = blockIdx.x;
    __bf16* d = DS + (size_t)row * CPAD;
    float m = mu[row], rs = rsd[row];
    float s = 0.f;
    for (int c = threadIdx.x; c < CPAD; c += 256) {
        float p = 0.f;
        if (c < CREAL)
            p = __expf(-LAMDA * tanhf((bf2f(d[c]) - m) * rs));
        d[c] = f2bf(p);
        s += p;
    }
    __shared__ float ss[256];
    ss[threadIdx.x] = s;
    __syncthreads();
    for (int st = 128; st > 0; st >>= 1) {
        if (threadIdx.x < st) ss[threadIdx.x] += ss[threadIdx.x + st];
        __syncthreads();
    }
    if (threadIdx.x == 0) inv9[row] = 1.0f / (ss[0] * 9.0f);
}

// ---------------- GEMM2: S = softmax(CA) @ patch1 (bf16 WMMA), scaled by inv9 ----------------
__global__ __launch_bounds__(128) void gemm2_kernel(const __bf16* __restrict__ CA,
                                                    const __bf16* __restrict__ patch1T,
                                                    const float* __restrict__ inv9,
                                                    float* __restrict__ S) {
    int blk = blockIdx.x;                // b*256 + mtile
    int b = blk >> 8, mt = blk & 255;
    int tid = threadIdx.x, lane = tid & 31, wv = tid >> 5;
    int r = lane & 15, kh = lane >> 4;

    v8f acc[9];
#pragma unroll
    for (int q = 0; q < 9; ++q)
#pragma unroll
        for (int v = 0; v < 8; ++v) acc[q][v] = 0.f;

    const __bf16* arow = CA + (size_t)(b * MTOT + mt * 16 + r) * CPAD + kh * 8;
    const __bf16* ptb  = patch1T + (size_t)b * PDIM * CPAD + (size_t)kh * 8;

    for (int kt = 0; kt < CPAD / 32; ++kt) {   // 121 k-steps, 9 WMMAs each
        v8bf alo = *(const v8bf*)(arow + kt * 32);
        v8bf ahi = *(const v8bf*)(arow + kt * 32 + 16);
        v16bf av = joinbf(alo, ahi);
        __builtin_prefetch(arow + (kt + 1) * 32, 0, 1);
#pragma unroll
        for (int q = 0; q < 9; ++q) {
            int n = (wv * 9 + q) * 16 + r;
            const __bf16* brow = ptb + (size_t)n * CPAD + kt * 32;
            v8bf blo = *(const v8bf*)brow;
            v8bf bhi = *(const v8bf*)(brow + 16);
            acc[q] = __builtin_amdgcn_wmma_f32_16x16x32_bf16(false, av,
                                                             false, joinbf(blo, bhi),
                                                             (short)0, acc[q], false, false);
        }
    }

    float scl[8];
#pragma unroll
    for (int v = 0; v < 8; ++v)
        scl[v] = inv9[b * MTOT + mt * 16 + v + 8 * kh];
    float* Sb = S + (size_t)(b * MTOT + mt * 16) * PDIM;
#pragma unroll
    for (int q = 0; q < 9; ++q) {
        int n = (wv * 9 + q) * 16 + r;
#pragma unroll
        for (int v = 0; v < 8; ++v)
            Sb[(size_t)(v + 8 * kh) * PDIM + n] = acc[q][v] * scl[v];
    }
}

// ---------------- ACL scatter (transposed-conv taps) + mask blend + concat ----------------
__global__ __launch_bounds__(256) void acl_concat_kernel(const float* __restrict__ bg_in,
                                                         const float* __restrict__ mask,
                                                         const float* __restrict__ S,
                                                         __bf16* __restrict__ con1) {
    int idx = blockIdx.x * 256 + threadIdx.x;      // B*4096*64
    if (idx >= BB * MTOT * DD) return;
    int ch = idx & 63;
    int m  = (idx >> 6) & 4095;
    int b  = idx >> 18;
    int y = m >> 6, x = m & 63;
    float acl = 0.f;
    const float* Sb = S + (size_t)b * MTOT * PDIM;
#pragma unroll
    for (int i = 0; i < 3; ++i) {
        int yy = y + 1 - i;
        if (yy < 0 || yy >= HH) continue;
#pragma unroll
        for (int j = 0; j < 3; ++j) {
            int xx = x + 1 - j;
            if (xx < 0 || xx >= WW) continue;
            acl += Sb[(size_t)(yy * WW + xx) * PDIM + (i * 3 + j) * DD + ch];
        }
    }
    float m0   = mask[((size_t)(b * HH + y) * WW + x) * DD];
    float bgin = bg_in[idx];
    float bgv  = bgin * m0;
    float aclf = bgv + acl * (1.f - m0);
    size_t base = (size_t)(b * MTOT + m) * 128;
    con1[base + ch]      = f2bf(bgin);
    con1[base + 64 + ch] = f2bf(aclf);
}

// ---------------- final 1x1 conv (M=8192,K=128,N=64) + bias + ELU via WMMA ----------------
__global__ __launch_bounds__(32) void ml_kernel(const __bf16* __restrict__ con1,
                                                const __bf16* __restrict__ wt,
                                                const float* __restrict__ bias,
                                                float* __restrict__ out) {
    int mt = blockIdx.x;                 // 512 tiles of 16 rows
    int lane = threadIdx.x;
    int r = lane & 15, kh = lane >> 4;
    v8f acc[4];
#pragma unroll
    for (int nt = 0; nt < 4; ++nt)
#pragma unroll
        for (int v = 0; v < 8; ++v) acc[nt][v] = 0.f;

    const __bf16* arow = con1 + (size_t)(mt * 16 + r) * 128 + kh * 8;
#pragma unroll
    for (int kt = 0; kt < 4; ++kt) {
        v8bf alo = *(const v8bf*)(arow + kt * 32);
        v8bf ahi = *(const v8bf*)(arow + kt * 32 + 16);
        v16bf av = joinbf(alo, ahi);
#pragma unroll
        for (int nt = 0; nt < 4; ++nt) {
            int n = nt * 16 + r;
            const __bf16* brow = wt + (size_t)n * 128 + kh * 8 + kt * 32;
            v8bf blo = *(const v8bf*)brow;
            v8bf bhi = *(const v8bf*)(brow + 16);
            acc[nt] = __builtin_amdgcn_wmma_f32_16x16x32_bf16(false, av,
                                                              false, joinbf(blo, bhi),
                                                              (short)0, acc[nt], false, false);
        }
    }
#pragma unroll
    for (int nt = 0; nt < 4; ++nt) {
        int n = nt * 16 + r;
        float bz = bias[n];
#pragma unroll
        for (int v = 0; v < 8; ++v) {
            int ml = v + 8 * kh;
            float o = acc[nt][v] + bz;
            out[(size_t)(mt * 16 + ml) * 64 + n] = (o > 0.f) ? o : (__expf(o) - 1.f);
        }
    }
}

// ---------------- host-side orchestration ----------------
extern "C" void kernel_launch(void* const* d_in, const int* in_sizes, int n_in,
                              void* d_out, int out_size, void* d_ws, size_t ws_size,
                              hipStream_t stream) {
    const float* bg_in = (const float*)d_in[0];
    const float* fg_in = (const float*)d_in[1];
    const float* mask  = (const float*)d_in[2];
    const float* W_ml  = (const float*)d_in[3];
    const float* b_ml  = (const float*)d_in[4];
    float* out = (float*)d_out;

    char* ws = (char*)d_ws;
    size_t off = 0;
    auto alloc = [&](size_t bytes) -> char* {
        char* p = ws + off;
        off = (off + bytes + 255) & ~(size_t)255;
        return p;
    };
    __bf16* patch2  = (__bf16*)alloc((size_t)BB * MTOT * PDIM * 2);
    __bf16* patch1  = (__bf16*)alloc((size_t)BB * CPAD * PDIM * 2);
    __bf16* patch1T = (__bf16*)alloc((size_t)BB * PDIM * CPAD * 2);
    float*  k1d     = (float*)alloc((size_t)BB * CPAD * 4);
    float*  wwd     = (float*)alloc((size_t)BB * MTOT * 4);
    __bf16* DS      = (__bf16*)alloc((size_t)BB * MTOT * CPAD * 2);  // DS1, then CA in place
    float*  mu      = (float*)alloc((size_t)BB * MTOT * 4);
    float*  rsd     = (float*)alloc((size_t)BB * MTOT * 4);
    float*  inv9    = (float*)alloc((size_t)BB * MTOT * 4);
    float*  S       = (float*)alloc((size_t)BB * MTOT * PDIM * 4);
    __bf16* con1    = (__bf16*)alloc((size_t)BB * MTOT * 128 * 2);
    __bf16* wt      = (__bf16*)alloc((size_t)64 * 128 * 2);

    prep_fg_kernel<<<BB * MTOT, 64, 0, stream>>>(fg_in, patch2, wwd);
    prep_bg_kernel<<<BB * CPAD, 64, 0, stream>>>(bg_in, mask, patch1, patch1T, k1d);
    prep_w_kernel<<<(64 * 128 + 255) / 256, 256, 0, stream>>>(W_ml, wt);

    gemm1_kernel<<<dim3(BB * 256, (CPAD / 16 + 3) / 4), 128, 0, stream>>>(
        patch2, patch1, k1d, wwd, DS);
    stats_kernel<<<BB * MTOT, 256, 0, stream>>>(DS, mu, rsd);
    expca_kernel<<<BB * MTOT, 256, 0, stream>>>(DS, mu, rsd, inv9);
    gemm2_kernel<<<BB * 256, 128, 0, stream>>>(DS, patch1T, inv9, S);

    acl_concat_kernel<<<(BB * MTOT * DD + 255) / 256, 256, 0, stream>>>(
        bg_in, mask, S, con1);
    ml_kernel<<<BB * MTOT / 16, 32, 0, stream>>>(con1, wt, b_ml, out);
}
